// MultiAtt_23854248362452
// MI455X (gfx1250) — compile-verified, matrix operand order checked by reading, not run on hardware
//
#include <hip/hip_runtime.h>

typedef _Float16 f16;
typedef _Float16 v16h __attribute__((ext_vector_type(16)));
typedef float v8f __attribute__((ext_vector_type(8)));

#define WMMA_F16(a, b, c) \
  __builtin_amdgcn_wmma_f32_16x16x32_f16(false, (a), false, (b), (short)0, (c), false, false)

// Load one wave32 16x16x32 f16 A/B fragment from a row-major matrix.
// Lane l holds row (l&15); K chunks [ks..ks+7] and [ks+16..ks+23], ks = 8*(l>>4).
// p must point at (row_base + k0); ks passed per-lane.
__device__ __forceinline__ v16h frag_load(const f16* p, int ks) {
  union { v16h h; uint4 u[2]; } f;
  f.u[0] = *(const uint4*)(p + ks);
  f.u[1] = *(const uint4*)(p + ks + 16);
  return f.h;
}

// ---------------------------------------------------------------------------
// Generic batched GEMM: C[b] (MxN, f32) = A[b] (MxK f16 row-major) * BT[b]^T
// where BT is stored as (N x K) f16 row-major. M,N mult of 16, K mult of 32.
// One wave per 16x16 output tile.
// ---------------------------------------------------------------------------
__global__ __launch_bounds__(256) void gemm_f16_wmma(
    const f16* __restrict__ A, long sAb, int lda,
    const f16* __restrict__ BT, long sBb, int ldb,
    float* __restrict__ C, long sCb, int ldc,
    int M, int N, int K)
{
  int lane = threadIdx.x & 31;
  int wave = threadIdx.x >> 5;
  int tilesN = N >> 4;
  int tilesM = M >> 4;
  int tile = blockIdx.x * (blockDim.x >> 5) + wave;
  if (tile >= tilesM * tilesN) return;           // wave-uniform exit
  int tm = (tile / tilesN) << 4;
  int tn = (tile % tilesN) << 4;
  int b = blockIdx.y;
  const f16* Ab = A + (long)b * sAb;
  const f16* Bb = BT + (long)b * sBb;
  float* Cb = C + (long)b * sCb;

  int r = lane & 15, hi = lane >> 4, ks = hi * 8;
  const f16* arow = Ab + (long)(tm + r) * lda;
  const f16* brow = Bb + (long)(tn + r) * ldb;

  v8f acc = {0.f, 0.f, 0.f, 0.f, 0.f, 0.f, 0.f, 0.f};
  for (int k0 = 0; k0 < K; k0 += 32) {
    v16h a = frag_load(arow + k0, ks);
    v16h bf = frag_load(brow + k0, ks);
    acc = WMMA_F16(a, bf, acc);
  }
  // C/D layout: lane l holds column n = (l&15); VGPR i -> row tm + i + 8*hi.
  float* cp = Cb + (long)(tm + hi * 8) * ldc + tn + r;
#pragma unroll
  for (int i = 0; i < 8; ++i) cp[(long)i * ldc] = acc[i];
}

// ---------------------------------------------------------------------------
// Fused dot-attention scores (dominant kernel, ~137 GFLOP per call):
//   s[b,m,n] = sum_h vd[h] * tanh( sum_d x[b,n,d]*y[b,m,d]*Wd[d,h] )
// One block per (b,m); 8 waves, wave w owns n-tile w. WdT is H x D f16.
// ---------------------------------------------------------------------------
__global__ __launch_bounds__(256) void dot_score_wmma(
    const f16* __restrict__ xh,   // [B,128,1024]
    const f16* __restrict__ yh,   // [B,128,1024]
    const f16* __restrict__ WdT,  // [512,1024]
    const float* __restrict__ vd, // [512]
    float* __restrict__ s)        // [B,128,128]
{
  const int N = 128, D = 1024, H = 512;
  int b = blockIdx.y, m = blockIdx.x;

  __shared__ f16 yrow[D];
  const f16* yp = yh + ((long)b * N + m) * D;
  for (int i = threadIdx.x; i < D / 8; i += blockDim.x)
    ((uint4*)yrow)[i] = ((const uint4*)yp)[i];
  __syncthreads();

  int lane = threadIdx.x & 31, wave = threadIdx.x >> 5;
  int r = lane & 15, hi = lane >> 4, ks = hi * 8;
  int n0 = wave * 16;
  const f16* xrow = xh + ((long)b * N + n0 + r) * D;

  float rowsum[8];
#pragma unroll
  for (int i = 0; i < 8; ++i) rowsum[i] = 0.f;

  for (int hc = 0; hc < H; hc += 64) {
    v8f acc[4];
#pragma unroll
    for (int j = 0; j < 4; ++j)
      acc[j] = (v8f){0.f, 0.f, 0.f, 0.f, 0.f, 0.f, 0.f, 0.f};

    const f16* wbase = WdT + (long)(hc + r) * D;
    for (int k0 = 0; k0 < D; k0 += 32) {
      v16h xa = frag_load(xrow + k0, ks);
      v16h ya = frag_load(yrow + k0, ks);   // LDS reads
      v16h a = xa * ya;                     // packed f16 elementwise product
#pragma unroll
      for (int j = 0; j < 4; ++j) {
        v16h bf = frag_load(wbase + (long)(j * 16) * D + k0, ks);
        acc[j] = WMMA_F16(a, bf, acc[j]);
      }
    }
    // tanh + weight by vd[h]; lane's column is h = hc + j*16 + r.
#pragma unroll
    for (int j = 0; j < 4; ++j) {
      float vdv = vd[hc + j * 16 + r];
#pragma unroll
      for (int i = 0; i < 8; ++i)
        rowsum[i] += tanhf(acc[j][i]) * vdv;
    }
  }
  // Reduce across the 16 lanes holding different h columns (same row set).
#pragma unroll
  for (int off = 1; off < 16; off <<= 1) {
#pragma unroll
    for (int i = 0; i < 8; ++i)
      rowsum[i] += __shfl_xor(rowsum[i], off, 32);
  }
  if (r == 0) {
    float* sp = s + ((long)b * N + m) * N + n0 + hi * 8;
#pragma unroll
    for (int i = 0; i < 8; ++i) sp[i] = rowsum[i];
  }
}

// ---------------------------------------------------------------------------
// Small f32 kernels for the cheap stages
// ---------------------------------------------------------------------------
__global__ void cvt_f32_f16(const float* __restrict__ in, f16* __restrict__ out, long n) {
  long i = (long)blockIdx.x * blockDim.x + threadIdx.x;
  if (i < n) out[i] = (f16)in[i];
}

// WT[n][k] = (f16) W[k][n];  W is K x Ncols row-major.
__global__ void transpose_cvt(const float* __restrict__ W, f16* __restrict__ WT,
                              int K, int Ncols) {
  int k = blockIdx.x * blockDim.x + threadIdx.x;
  int n = blockIdx.y;
  if (k < K) WT[(long)n * K + k] = (f16)W[(long)k * Ncols + n];
}

// s[b,m,n] = sum_h v[h] * tanh(Arow[b,n,h] + sign*Brow[b,m,h])
__global__ void addsub_score(const float* __restrict__ Arow, const float* __restrict__ Brow,
                             const float* __restrict__ v, float* __restrict__ s, float sign) {
  const int N = 128, H = 512;
  int b = blockIdx.y, m = blockIdx.x;
  __shared__ float bsh[512];
  __shared__ float vsh[512];
  const float* bp = Brow + ((long)b * N + m) * H;
  for (int i = threadIdx.x; i < H; i += blockDim.x) {
    bsh[i] = sign * bp[i];
    vsh[i] = v[i];
  }
  __syncthreads();
  for (int n = threadIdx.x; n < N; n += blockDim.x) {
    const float* ap = Arow + ((long)b * N + n) * H;
    float acc = 0.f;
    for (int h = 0; h < H; ++h) acc += vsh[h] * tanhf(ap[h] + bsh[h]);
    s[((long)b * N + m) * N + n] = acc;
  }
}

// In-place softmax over rows of length 128; blockIdx.x = row, 128 threads.
__global__ void softmax128(float* __restrict__ s) {
  float* p = s + (long)blockIdx.x * 128;
  int t = threadIdx.x;
  __shared__ float red[128];
  float v = p[t];
  red[t] = v;
  __syncthreads();
  for (int off = 64; off; off >>= 1) {
    if (t < off) red[t] = fmaxf(red[t], red[t + off]);
    __syncthreads();
  }
  float mx = red[0];
  __syncthreads();
  float e = expf(v - mx);
  red[t] = e;
  __syncthreads();
  for (int off = 64; off; off >>= 1) {
    if (t < off) red[t] += red[t + off];
    __syncthreads();
  }
  p[t] = e / red[0];
}

// qt[b,m,d] = sum_n p[b,m,n] * x[b,n,d]
__global__ void attn_context(const float* __restrict__ p, const float* __restrict__ x,
                             float* __restrict__ qt) {
  const int N = 128, D = 1024;
  int d = blockIdx.x * blockDim.x + threadIdx.x;
  int m = blockIdx.y, b = blockIdx.z;
  const float* pr = p + ((long)b * N + m) * N;
  const float* xb = x + (long)b * N * D + d;
  float acc = 0.f;
  for (int n = 0; n < N; ++n) acc += pr[n] * xb[(long)n * D];
  qt[((long)b * N + m) * D + d] = acc;
}

// agg[b,m,0:1024] = x1; agg[b,m,1024:2048] = x1 (max identity seed).
__global__ void agg_init(const float* __restrict__ x1, float* __restrict__ agg) {
  int d = blockIdx.x * blockDim.x + threadIdx.x;
  int m = blockIdx.y, b = blockIdx.z;
  float v = x1[((long)b * 128 + m) * 1024 + d];
  long o = ((long)b * 128 + m) * 2048;
  agg[o + d] = v;
  agg[o + 1024 + d] = v;
}

// agg[b,m,1024+d] = max(agg[b,m,1024+d], qt[b,m,d])
__global__ void fold_max(const float* __restrict__ qt, float* __restrict__ agg) {
  int d = blockIdx.x * blockDim.x + threadIdx.x;
  int m = blockIdx.y, b = blockIdx.z;
  long src = ((long)b * 128 + m) * 1024 + d;
  long dst = ((long)b * 128 + m) * 2048 + 1024 + d;
  agg[dst] = fmaxf(agg[dst], qt[src]);
}

// out[row] = sum_h v[h]*tanh(T[row,h])
__global__ void tanh_dot(const float* __restrict__ T, const float* __restrict__ v,
                         float* __restrict__ out, int rows, int H) {
  int rIdx = blockIdx.x * blockDim.x + threadIdx.x;
  if (rIdx >= rows) return;
  const float* t = T + (long)rIdx * H;
  float acc = 0.f;
  for (int h = 0; h < H; ++h) acc += v[h] * tanhf(t[h]);
  out[rIdx] = acc;
}

// out[b,d] = sum_n p[b,n] * x[b,n,d]  (rows of 128)
__global__ void weighted_sum_rows(const float* __restrict__ p, const float* __restrict__ x,
                                  float* __restrict__ out, int Dd) {
  int d = blockIdx.x * blockDim.x + threadIdx.x;
  int b = blockIdx.y;
  if (d >= Dd) return;
  const float* pr = p + (long)b * 128;
  const float* xb = x + (long)b * 128 * Dd + d;
  float acc = 0.f;
  for (int n = 0; n < 128; ++n) acc += pr[n] * xb[(long)n * Dd];
  out[(long)b * Dd + d] = acc;
}

// out[b,h] = sum_d rq[b,d] * W[d,h]
__global__ void small_matvec(const float* __restrict__ rq, const float* __restrict__ W,
                             float* __restrict__ out, int Din, int Dout) {
  int h = blockIdx.x * blockDim.x + threadIdx.x;
  int b = blockIdx.y;
  if (h >= Dout) return;
  const float* r = rq + (long)b * Din;
  float acc = 0.f;
  for (int d = 0; d < Din; ++d) acc += r[d] * W[(long)d * Dout + h];
  out[(long)b * Dout + h] = acc;
}

// sm[b,m] = sum_h vp[h] * (traw[b,m,h] + rqp[b,h])
__global__ void t_score(const float* __restrict__ traw, const float* __restrict__ rqp,
                        const float* __restrict__ vp, float* __restrict__ sm) {
  int m = threadIdx.x;
  int b = blockIdx.x;
  const float* t = traw + ((long)b * 128 + m) * 512;
  const float* rp = rqp + (long)b * 512;
  float acc = 0.f;
  for (int h = 0; h < 512; ++h) acc += vp[h] * (t[h] + rp[h]);
  sm[b * 128 + m] = acc;
}

// out[b,j] = relu(sum_d rp[b,d]*Wpred[d,j] + bpred[j]), 8x2 outputs
__global__ void pred_kernel(const float* __restrict__ rp, const float* __restrict__ Wpred,
                            const float* __restrict__ bpred, float* __restrict__ out) {
  int tid = threadIdx.x;
  if (tid >= 16) return;
  int j = tid & 1, b = tid >> 1;
  const float* r = rp + (long)b * 2048;
  float acc = bpred[j];
  for (int d = 0; d < 2048; ++d) acc += r[d] * Wpred[d * 2 + j];
  out[b * 2 + j] = fmaxf(acc, 0.f);
}

// ---------------------------------------------------------------------------
extern "C" void kernel_launch(void* const* d_in, const int* in_sizes, int n_in,
                              void* d_out, int out_size, void* d_ws, size_t ws_size,
                              hipStream_t stream) {
  (void)in_sizes; (void)n_in; (void)out_size; (void)ws_size;
  const int B = 8, N = 128, D = 1024, H = 512;

  const float* x0   = (const float*)d_in[0];
  const float* x1   = (const float*)d_in[1];
  const float* Wc1  = (const float*)d_in[2];
  const float* Wc2  = (const float*)d_in[3];
  const float* vc   = (const float*)d_in[4];
  const float* Wb   = (const float*)d_in[5];
  const float* Wd1  = (const float*)d_in[6];
  const float* vd1  = (const float*)d_in[7];
  const float* Wd2  = (const float*)d_in[8];
  const float* vd2  = (const float*)d_in[9];
  const float* Wm   = (const float*)d_in[10];
  const float* vm   = (const float*)d_in[11];
  const float* Wq   = (const float*)d_in[12];
  const float* vq   = (const float*)d_in[13];
  const float* Wp1  = (const float*)d_in[14];
  const float* Wp2  = (const float*)d_in[15];
  const float* vp   = (const float*)d_in[16];
  const float* Wpred = (const float*)d_in[17];
  const float* bpred = (const float*)d_in[18];
  float* out = (float*)d_out;

  char* ws = (char*)d_ws;
  size_t off = 0;
  auto alloc = [&](size_t bytes) -> void* {
    void* p = ws + off;
    off = (off + bytes + 255) & ~(size_t)255;
    return p;
  };

  const long XN = (long)B * N * D;          // 1,048,576
  f16* x0h    = (f16*)alloc(XN * 2);
  f16* x1h    = (f16*)alloc(XN * 2);
  f16* WcT1h  = (f16*)alloc((long)H * D * 2);
  f16* WcT2h  = (f16*)alloc((long)H * D * 2);
  f16* WdT1h  = (f16*)alloc((long)H * D * 2);
  f16* WdT2h  = (f16*)alloc((long)H * D * 2);
  f16* WmTh   = (f16*)alloc((long)H * D * 2);
  f16* WqTh   = (f16*)alloc((long)H * D * 2);
  f16* WbTh   = (f16*)alloc((long)D * D * 2);
  f16* Wp1Th  = (f16*)alloc((long)H * 2 * D * 2);
  float* Ac   = (float*)alloc((long)B * N * H * 4);
  float* Bc   = (float*)alloc((long)B * N * H * 4);
  float* Am   = (float*)alloc((long)B * N * H * 4);
  float* Bm   = (float*)alloc((long)B * N * H * 4);
  float* Tq   = (float*)alloc((long)B * N * H * 4);
  float* XB   = (float*)alloc(XN * 4);
  f16*   XBh  = (f16*)alloc(XN * 2);
  float* sbuf = (float*)alloc((long)B * N * N * 4);
  float* qt   = (float*)alloc(XN * 4);
  float* agg  = (float*)alloc((long)B * N * 2 * D * 4);
  f16*   aggh = (f16*)alloc((long)B * N * 2 * D * 2);
  float* traw = (float*)alloc((long)B * N * H * 4);
  float* sj   = (float*)alloc((long)B * N * 4);
  float* rq   = (float*)alloc((long)B * D * 4);
  float* rqp  = (float*)alloc((long)B * H * 4);
  float* smv  = (float*)alloc((long)B * N * 4);
  float* rpv  = (float*)alloc((long)B * 2 * D * 4);

  auto gemm = [&](const f16* A, long sAb, int lda, const f16* BT, long sBb, int ldb,
                  float* C, long sCb, int ldc, int M, int Nn, int K, int batch) {
    int tiles = (M / 16) * (Nn / 16);
    int blocks = (tiles + 7) / 8;
    gemm_f16_wmma<<<dim3(blocks, batch), dim3(256), 0, stream>>>(
        A, sAb, lda, BT, sBb, ldb, C, sCb, ldc, M, Nn, K);
  };

  // ---- precision conversion + weight transposes ----
  cvt_f32_f16<<<dim3((XN + 255) / 256), dim3(256), 0, stream>>>(x0, x0h, XN);
  cvt_f32_f16<<<dim3((XN + 255) / 256), dim3(256), 0, stream>>>(x1, x1h, XN);
  transpose_cvt<<<dim3(D / 256, H), dim3(256), 0, stream>>>(Wc1, WcT1h, D, H);
  transpose_cvt<<<dim3(D / 256, H), dim3(256), 0, stream>>>(Wc2, WcT2h, D, H);
  transpose_cvt<<<dim3(D / 256, H), dim3(256), 0, stream>>>(Wd1, WdT1h, D, H);
  transpose_cvt<<<dim3(D / 256, H), dim3(256), 0, stream>>>(Wd2, WdT2h, D, H);
  transpose_cvt<<<dim3(D / 256, H), dim3(256), 0, stream>>>(Wm, WmTh, D, H);
  transpose_cvt<<<dim3(D / 256, H), dim3(256), 0, stream>>>(Wq, WqTh, D, H);
  transpose_cvt<<<dim3(D / 256, D), dim3(256), 0, stream>>>(Wb, WbTh, D, D);
  transpose_cvt<<<dim3(2 * D / 256, H), dim3(256), 0, stream>>>(Wp1, Wp1Th, 2 * D, H);

  // ---- projections (WMMA) ----
  gemm(x0h, (long)N * D, D, WcT1h, 0, D, Ac, (long)N * H, H, N, H, D, B);  // x0@Wc1
  gemm(x1h, (long)N * D, D, WcT2h, 0, D, Bc, (long)N * H, H, N, H, D, B);  // x1@Wc2
  gemm(x0h, (long)N * D, D, WmTh, 0, D, Am, (long)N * H, H, N, H, D, B);   // x0@Wm
  gemm(x1h, (long)N * D, D, WmTh, 0, D, Bm, (long)N * H, H, N, H, D, B);   // x1@Wm
  gemm(x0h, (long)N * D, D, WqTh, 0, D, Tq, (long)N * H, H, N, H, D, B);   // x0@Wq
  gemm(x0h, (long)N * D, D, WbTh, 0, D, XB, (long)N * D, D, N, D, D, B);   // x0@Wb
  cvt_f32_f16<<<dim3((XN + 255) / 256), dim3(256), 0, stream>>>(XB, XBh, XN);

  // ---- aggregation seed: agg = concat(x1, max-identity=x1) ----
  agg_init<<<dim3(D / 256, N, B), dim3(256), 0, stream>>>(x1, agg);

  // ---- attention 1: concat ----
  addsub_score<<<dim3(N, B), dim3(128), 0, stream>>>(Ac, Bc, vc, sbuf, 1.0f);
  softmax128<<<dim3(B * N), dim3(128), 0, stream>>>(sbuf);
  attn_context<<<dim3(D / 256, N, B), dim3(256), 0, stream>>>(sbuf, x0, qt);
  fold_max<<<dim3(D / 256, N, B), dim3(256), 0, stream>>>(qt, agg);

  // ---- attention 2: bilinear  s = x1 . (x0@Wb)^T ----
  gemm(x1h, (long)N * D, D, XBh, (long)N * D, D, sbuf, (long)N * N, N, N, N, D, B);
  softmax128<<<dim3(B * N), dim3(128), 0, stream>>>(sbuf);
  attn_context<<<dim3(D / 256, N, B), dim3(256), 0, stream>>>(sbuf, x0, qt);
  fold_max<<<dim3(D / 256, N, B), dim3(256), 0, stream>>>(qt, agg);

  // ---- attention 3: dot (x=x0, y=x1, Wd1) ----
  dot_score_wmma<<<dim3(N, B), dim3(256), 0, stream>>>(x0h, x1h, WdT1h, vd1, sbuf);
  softmax128<<<dim3(B * N), dim3(128), 0, stream>>>(sbuf);
  attn_context<<<dim3(D / 256, N, B), dim3(256), 0, stream>>>(sbuf, x0, qt);
  fold_max<<<dim3(D / 256, N, B), dim3(256), 0, stream>>>(qt, agg);

  // ---- attention 4: dot swapped (x=x1, y=x0, Wd2); context over x1 ----
  dot_score_wmma<<<dim3(N, B), dim3(256), 0, stream>>>(x1h, x0h, WdT2h, vd2, sbuf);
  softmax128<<<dim3(B * N), dim3(128), 0, stream>>>(sbuf);
  attn_context<<<dim3(D / 256, N, B), dim3(256), 0, stream>>>(sbuf, x1, qt);
  fold_max<<<dim3(D / 256, N, B), dim3(256), 0, stream>>>(qt, agg);

  // ---- attention 5: minus ----
  addsub_score<<<dim3(N, B), dim3(128), 0, stream>>>(Am, Bm, vm, sbuf, -1.0f);
  softmax128<<<dim3(B * N), dim3(128), 0, stream>>>(sbuf);
  attn_context<<<dim3(D / 256, N, B), dim3(256), 0, stream>>>(sbuf, x0, qt);
  fold_max<<<dim3(D / 256, N, B), dim3(256), 0, stream>>>(qt, agg);

  // ---- rq path: sj = (tanh(x0@Wq)).vq, softmax over n, rq = sum p*x0 ----
  tanh_dot<<<dim3((B * N + 255) / 256), dim3(256), 0, stream>>>(Tq, vq, sj, B * N, H);
  softmax128<<<dim3(B), dim3(128), 0, stream>>>(sj);
  weighted_sum_rows<<<dim3(D / 256, B), dim3(256), 0, stream>>>(sj, x0, rq, D);

  // ---- aggregation layer ----
  long aggN = (long)B * N * 2 * D;
  cvt_f32_f16<<<dim3((int)((aggN + 255) / 256)), dim3(256), 0, stream>>>(agg, aggh, aggN);
  gemm(aggh, (long)N * 2 * D, 2 * D, Wp1Th, 0, 2 * D, traw, (long)N * H, H, N, H, 2 * D, B);
  small_matvec<<<dim3(H / 256, B), dim3(256), 0, stream>>>(rq, Wp2, rqp, D, H);
  t_score<<<dim3(B), dim3(128), 0, stream>>>(traw, rqp, vp, smv);
  softmax128<<<dim3(B), dim3(128), 0, stream>>>(smv);
  weighted_sum_rows<<<dim3(2 * D / 256, B), dim3(256), 0, stream>>>(smv, agg, rpv, 2 * D);
  pred_kernel<<<dim3(1), dim3(16), 0, stream>>>(rpv, Wpred, bpred, out);
}
